// MultiHeadAttention_5892695130133
// MI455X (gfx1250) — compile-verified, matrix operand order checked by reading, not run on hardware
//
#include <hip/hip_runtime.h>

// ---------------------------------------------------------------------------
// MHA forward for MI455X (gfx1250): bf16 WMMA everywhere, f32 accumulation.
// Pipeline: cvt(f32->bf16) -> QKV GEMMs -> flash attention -> out GEMM+bias.
// GEMM: 128x128 block, 4 waves of 64x64 (16 WMMA / 8 frag-loads per k-step),
// double-buffered LDS, async global->LDS copies when the builtin exists.
// ---------------------------------------------------------------------------

typedef __bf16 bf16_t;
typedef __attribute__((ext_vector_type(16))) __bf16 v16bf;
typedef __attribute__((ext_vector_type(8)))  __bf16 v8bf;
typedef __attribute__((ext_vector_type(8)))  float  v8f;
typedef __attribute__((ext_vector_type(4)))  int    v4i;

#define B_  2
#define S_  2048
#define D_  1024
#define H_  16
#define DK_ 64
#define M_  (B_ * S_)      // 4096 tokens

// ---- CDNA5 async global->LDS copy (ASYNCcnt path), with safe fallback ----
#if defined(__has_builtin)
#  if __has_builtin(__builtin_amdgcn_global_load_async_to_lds_b128) && \
      __has_builtin(__builtin_amdgcn_s_wait_asynccnt)
#    define HAVE_ASYNC 1
#  endif
#endif
#ifndef HAVE_ASYNC
#  define HAVE_ASYNC 0
#endif

#if HAVE_ASYNC
__device__ inline void cp16_async(const bf16_t* g, bf16_t* l) {
  // Param types (from clang-22 diagnostic): (v4i AS1*, v4i AS3*, imm, imm).
  // Integer casts: AS1 value is the flat value; flat LDS addresses carry the
  // LDS offset in the low 32 bits (ISA 10.2), so truncation is the AS3 offset.
  __attribute__((address_space(1))) v4i* gp =
      (__attribute__((address_space(1))) v4i*)(unsigned long long)g;
  __attribute__((address_space(3))) v4i* lp =
      (__attribute__((address_space(3))) v4i*)(unsigned int)(unsigned long long)l;
  __builtin_amdgcn_global_load_async_to_lds_b128(gp, lp, 0, 0);
}
__device__ inline void async_wait_all() { __builtin_amdgcn_s_wait_asynccnt(0); }
#else
__device__ inline void async_wait_all() {}
#endif

// ---------------- helpers ----------------

__device__ inline v8f vzero8() {
  v8f z = {0.f, 0.f, 0.f, 0.f, 0.f, 0.f, 0.f, 0.f};
  return z;
}

__device__ inline v8f wmma_bf16(v16bf a, v16bf b, v8f c) {
  return __builtin_amdgcn_wmma_f32_16x16x32_bf16(
      /*neg_a=*/false, a, /*neg_b=*/false, b,
      /*c_mod=*/(short)0, c, /*reuse_a=*/false, /*reuse_b=*/false);
}

// 16x32 bf16 fragment, ISA A layout (lane<16: K{0..7,16..23}; lane>=16:
// K{8..15,24..31}; row/col = lane&15). base = (row0,k0), stride ldh halves.
__device__ inline v16bf load_frag(const bf16_t* base, int ldh) {
  const int lane = threadIdx.x & 31;
  const bf16_t* p = base + (lane & 15) * ldh + ((lane >> 4) << 3);
  v8bf lo = *(const v8bf*)(p);
  v8bf hi = *(const v8bf*)(p + 16);
  return __builtin_shufflevector(lo, hi, 0, 1, 2, 3, 4, 5, 6, 7,
                                 8, 9, 10, 11, 12, 13, 14, 15);
}

// ---------------- f32 -> bf16 conversion ----------------

__global__ __launch_bounds__(256) void cvt_kernel(const float* __restrict__ src,
                                                  bf16_t* __restrict__ dst, int n) {
  int i = blockIdx.x * blockDim.x + threadIdx.x;
  int stride = gridDim.x * blockDim.x;
  for (; i < n; i += stride) dst[i] = (bf16_t)src[i];
}

// ---------------- GEMM: C[128x128] += A[128xK] * W[128xK]^T ----------------
// 128 threads = 4 waves in a 2x2 grid of 64x64 wave tiles; BK=32; LDS double
// buffered (stage stride BM*LDT halves).

#define BM 128
#define BN 128
#define BK 32
#define LDT 40           // 32 + 8 halves pad (80 B rows, 16 B aligned)
#define STG (BM * LDT)   // halves per stage

__device__ inline void load_stage(const bf16_t* __restrict__ A,
                                  const bf16_t* __restrict__ W, int Kdim,
                                  int blockM, int blockN, int k0,
                                  bf16_t* As, bf16_t* Bs) {
  const int tid = threadIdx.x;
#pragma unroll
  for (int i = 0; i < 4; ++i) {
    int seg = tid + i * 128;          // 512 segs of 8 halves per tile
    int row = seg >> 2;
    int c8  = (seg & 3) * 8;
    const bf16_t* ga = A + (size_t)(blockM + row) * Kdim + k0 + c8;
    const bf16_t* gb = W + (size_t)(blockN + row) * Kdim + k0 + c8;
#if HAVE_ASYNC
    cp16_async(ga, As + row * LDT + c8);
    cp16_async(gb, Bs + row * LDT + c8);
#else
    *(v8bf*)(As + row * LDT + c8) = *(const v8bf*)ga;
    *(v8bf*)(Bs + row * LDT + c8) = *(const v8bf*)gb;
    if (k0 + BK < Kdim) {
      __builtin_prefetch(ga + BK, 0, 1);
      __builtin_prefetch(gb + BK, 0, 1);
    }
#endif
  }
}

__device__ inline void compute_stage(const bf16_t* As, const bf16_t* Bs,
                                     int wm, int wn, v8f acc[4][4]) {
  v16bf af[4], bfr[4];
#pragma unroll
  for (int mf = 0; mf < 4; ++mf)
    af[mf] = load_frag(As + (wm + mf * 16) * LDT, LDT);
#pragma unroll
  for (int nf = 0; nf < 4; ++nf)
    bfr[nf] = load_frag(Bs + (wn + nf * 16) * LDT, LDT);
#pragma unroll
  for (int nf = 0; nf < 4; ++nf)
#pragma unroll
    for (int mf = 0; mf < 4; ++mf)
      acc[mf][nf] = wmma_bf16(af[mf], bfr[nf], acc[mf][nf]);
}

__device__ inline void gemm_main(const bf16_t* __restrict__ A,
                                 const bf16_t* __restrict__ W, int Kdim,
                                 v8f acc[4][4], bf16_t* As, bf16_t* Bs) {
  const int wave = threadIdx.x >> 5;
  const int wm = (wave >> 1) * 64;
  const int wn = (wave & 1) * 64;
  const int blockM = blockIdx.x * BM;
  const int blockN = blockIdx.y * BN;

  load_stage(A, W, Kdim, blockM, blockN, 0, As, Bs);
  async_wait_all();
  __syncthreads();

  int buf = 0;
  for (int k0 = 0; k0 < Kdim; k0 += BK) {
    int nxt = buf ^ 1;
    if (k0 + BK < Kdim)
      load_stage(A, W, Kdim, blockM, blockN, k0 + BK,
                 As + nxt * STG, Bs + nxt * STG);
    compute_stage(As + buf * STG, Bs + buf * STG, wm, wn, acc);
    async_wait_all();
    __syncthreads();
    buf = nxt;
  }
}

// QKV projection: out is bf16 [B, H, S, DK]
__global__ __launch_bounds__(128) void qkv_gemm_kernel(
    const bf16_t* __restrict__ A, const bf16_t* __restrict__ W,
    bf16_t* __restrict__ out) {
  __shared__ __align__(16) bf16_t As[2 * STG];
  __shared__ __align__(16) bf16_t Bs[2 * STG];
  v8f acc[4][4];
#pragma unroll
  for (int i = 0; i < 4; ++i)
#pragma unroll
    for (int j = 0; j < 4; ++j) acc[i][j] = vzero8();

  gemm_main(A, W, D_, acc, As, Bs);

  const int lane = threadIdx.x & 31;
  const int wave = threadIdx.x >> 5;
  const int wm = (wave >> 1) * 64, wn = (wave & 1) * 64;
  const int blockM = blockIdx.x * BM, blockN = blockIdx.y * BN;
#pragma unroll
  for (int mf = 0; mf < 4; ++mf) {
#pragma unroll
    for (int nf = 0; nf < 4; ++nf) {
      int nn = blockN + wn + nf * 16 + (lane & 15);
      int h = nn >> 6, dk = nn & 63;
#pragma unroll
      for (int j = 0; j < 8; ++j) {
        int m = blockM + wm + mf * 16 + ((lane >> 4) << 3) + j;
        int b = m >> 11, s = m & (S_ - 1);
        out[(((size_t)b * H_ + h) * S_ + s) * DK_ + dk] = (bf16_t)acc[mf][nf][j];
      }
    }
  }
}

// Output projection: f32 out = ctx @ Wo^T + bo, row-major [M_, D_]
__global__ __launch_bounds__(128) void out_gemm_kernel(
    const bf16_t* __restrict__ A, const bf16_t* __restrict__ W,
    const float* __restrict__ bias, float* __restrict__ out) {
  __shared__ __align__(16) bf16_t As[2 * STG];
  __shared__ __align__(16) bf16_t Bs[2 * STG];
  v8f acc[4][4];
#pragma unroll
  for (int i = 0; i < 4; ++i)
#pragma unroll
    for (int j = 0; j < 4; ++j) acc[i][j] = vzero8();

  gemm_main(A, W, D_, acc, As, Bs);

  const int lane = threadIdx.x & 31;
  const int wave = threadIdx.x >> 5;
  const int wm = (wave >> 1) * 64, wn = (wave & 1) * 64;
  const int blockM = blockIdx.x * BM, blockN = blockIdx.y * BN;
#pragma unroll
  for (int mf = 0; mf < 4; ++mf) {
#pragma unroll
    for (int nf = 0; nf < 4; ++nf) {
      int nn = blockN + wn + nf * 16 + (lane & 15);
      float bv = bias[nn];
#pragma unroll
      for (int j = 0; j < 8; ++j) {
        int m = blockM + wm + mf * 16 + ((lane >> 4) << 3) + j;
        out[(size_t)m * D_ + nn] = acc[mf][nf][j] + bv;
      }
    }
  }
}

// ---------------- flash attention ----------------
// grid: (S/128 query tiles, B*H).  256 threads = 8 waves; wave owns 16 q rows.
// KV blocks of 64 keys; K row-major in LDS, V transposed (dk-major).

#define KV  64
#define LDK 72   // 64 + 8 halves pad

__global__ __launch_bounds__(256) void flash_kernel(
    const bf16_t* __restrict__ Q, const bf16_t* __restrict__ Kq,
    const bf16_t* __restrict__ V, bf16_t* __restrict__ ctxout) {
  __shared__ __align__(16) bf16_t Ks[KV * LDK];        //  9 KB: [key][dk]
  __shared__ __align__(16) bf16_t Vt[DK_ * LDK];       //  9 KB: [dk][key]
  __shared__ __align__(16) bf16_t Ps[8 * 16 * LDK];    // 18 KB: wave-private P

  const int tid = threadIdx.x, lane = tid & 31, wave = tid >> 5;
  const int qb = blockIdx.x, bh = blockIdx.y;
  const int b = bh >> 4, h = bh & 15;
  const size_t head_off = (size_t)bh * S_ * DK_;
  const int q0 = qb * 128 + wave * 16;

  const bf16_t* qptr = Q + head_off + (size_t)q0 * DK_;
  v16bf aQ[2];
  aQ[0] = load_frag(qptr, DK_);
  aQ[1] = load_frag(qptr + 32, DK_);

  float rm[8], rs[8];
  alignas(32) float ctxf[4][8];
#pragma unroll
  for (int j = 0; j < 8; ++j) { rm[j] = -1e30f; rs[j] = 0.f; }
#pragma unroll
  for (int nt = 0; nt < 4; ++nt)
#pragma unroll
    for (int j = 0; j < 8; ++j) ctxf[nt][j] = 0.f;

  const float cexp = 0.125f * 1.44269504088896f;   // 1/sqrt(DK) * log2(e)
  const int nkv = (qb + 1) * 2;

  for (int kb = 0; kb < nkv; ++kb) {
#pragma unroll
    for (int i = 0; i < 2; ++i) {
      int seg = tid + i * 256;
      int row = seg >> 3;          // key index in block
      int c8  = (seg & 7) * 8;     // dk
      size_t g = head_off + (size_t)(kb * KV + row) * DK_ + c8;
      *(v8bf*)(Ks + row * LDK + c8) = *(const v8bf*)(Kq + g);
      v8bf vv = *(const v8bf*)(V + g);
#pragma unroll
      for (int e = 0; e < 8; ++e) Vt[(c8 + e) * LDK + row] = vv[e];
    }
    __syncthreads();

    alignas(32) float scf[4][8];
#pragma unroll
    for (int nt = 0; nt < 4; ++nt) {
      v8f sc = vzero8();
#pragma unroll
      for (int ks = 0; ks < 2; ++ks) {
        v16bf bK = load_frag(Ks + (nt * 16) * LDK + ks * 32, LDK);
        sc = wmma_bf16(aQ[ks], bK, sc);
      }
      *(v8f*)&scf[nt][0] = sc;
    }

    if (kb >= 2 * qb) {   // diagonal blocks: per-element causal mask
#pragma unroll
      for (int nt = 0; nt < 4; ++nt) {
        int key = kb * KV + nt * 16 + (lane & 15);
#pragma unroll
        for (int j = 0; j < 8; ++j) {
          int qg = q0 + ((lane >> 4) << 3) + j;
          if (key > qg) scf[nt][j] = -1e30f;
        }
      }
    }

    float bm[8];
#pragma unroll
    for (int j = 0; j < 8; ++j) bm[j] = -1e30f;
#pragma unroll
    for (int nt = 0; nt < 4; ++nt)
#pragma unroll
      for (int j = 0; j < 8; ++j) bm[j] = fmaxf(bm[j], scf[nt][j]);
#pragma unroll
    for (int j = 0; j < 8; ++j) {
#pragma unroll
      for (int mk = 1; mk < 16; mk <<= 1)
        bm[j] = fmaxf(bm[j], __shfl_xor(bm[j], mk, 32));
      float nm = fmaxf(rm[j], bm[j]);
      float alpha = exp2f((rm[j] - nm) * cexp);
      rm[j] = nm;
      rs[j] *= alpha;
#pragma unroll
      for (int nt = 0; nt < 4; ++nt) ctxf[nt][j] *= alpha;
    }

    float psum[8];
#pragma unroll
    for (int j = 0; j < 8; ++j) psum[j] = 0.f;
#pragma unroll
    for (int nt = 0; nt < 4; ++nt) {
#pragma unroll
      for (int j = 0; j < 8; ++j) {
        float p = exp2f((scf[nt][j] - rm[j]) * cexp);
        psum[j] += p;
        int mrow = ((lane >> 4) << 3) + j;
        Ps[(wave * 16 + mrow) * LDK + nt * 16 + (lane & 15)] = (bf16_t)p;
      }
    }
#pragma unroll
    for (int j = 0; j < 8; ++j) {
#pragma unroll
      for (int mk = 1; mk < 16; mk <<= 1)
        psum[j] += __shfl_xor(psum[j], mk, 32);
      rs[j] += psum[j];
    }

#pragma unroll
    for (int ks = 0; ks < 2; ++ks) {
      v16bf aP = load_frag(Ps + (wave * 16) * LDK + ks * 32, LDK);
#pragma unroll
      for (int nt = 0; nt < 4; ++nt) {
        v16bf bV = load_frag(Vt + (nt * 16) * LDK + ks * 32, LDK);
        v8f c = *(v8f*)&ctxf[nt][0];
        c = wmma_bf16(aP, bV, c);
        *(v8f*)&ctxf[nt][0] = c;
      }
    }
    __syncthreads();
  }

#pragma unroll
  for (int nt = 0; nt < 4; ++nt) {
#pragma unroll
    for (int j = 0; j < 8; ++j) {
      int sg = q0 + ((lane >> 4) << 3) + j;
      int dk = nt * 16 + (lane & 15);
      ctxout[((size_t)b * S_ + sg) * D_ + h * DK_ + dk] =
          (bf16_t)(ctxf[nt][j] / rs[j]);
    }
  }
}

// ---------------- launcher ----------------

extern "C" void kernel_launch(void* const* d_in, const int* in_sizes, int n_in,
                              void* d_out, int out_size, void* d_ws, size_t ws_size,
                              hipStream_t stream) {
  (void)in_sizes; (void)n_in; (void)out_size; (void)ws_size;
  const float* x  = (const float*)d_in[0];
  const float* Wq = (const float*)d_in[1];
  const float* Wk = (const float*)d_in[2];
  const float* Wv = (const float*)d_in[3];
  const float* Wo = (const float*)d_in[4];
  const float* bo = (const float*)d_in[5];
  float* out = (float*)d_out;

  char* ws = (char*)d_ws;                       // 48 MB used
  bf16_t* xb  = (bf16_t*)(ws);                  // [M_, D_]      8 MB
  bf16_t* Wqb = (bf16_t*)(ws + (8ull  << 20));  // [D_, D_]      2 MB
  bf16_t* Wkb = (bf16_t*)(ws + (10ull << 20));
  bf16_t* Wvb = (bf16_t*)(ws + (12ull << 20));
  bf16_t* Wob = (bf16_t*)(ws + (14ull << 20));
  bf16_t* Qb  = (bf16_t*)(ws + (16ull << 20));  // [B,H,S,DK]    8 MB
  bf16_t* Kb  = (bf16_t*)(ws + (24ull << 20));
  bf16_t* Vb  = (bf16_t*)(ws + (32ull << 20));
  bf16_t* Cb  = (bf16_t*)(ws + (40ull << 20));  // ctx [B,S,D]   8 MB

  cvt_kernel<<<1024, 256, 0, stream>>>(x,  xb,  M_ * D_);
  cvt_kernel<<<1024, 256, 0, stream>>>(Wq, Wqb, D_ * D_);
  cvt_kernel<<<1024, 256, 0, stream>>>(Wk, Wkb, D_ * D_);
  cvt_kernel<<<1024, 256, 0, stream>>>(Wv, Wvb, D_ * D_);
  cvt_kernel<<<1024, 256, 0, stream>>>(Wo, Wob, D_ * D_);

  dim3 gg(M_ / BM, D_ / BN);   // 32 x 8 blocks, 128 threads (4 wave32)
  qkv_gemm_kernel<<<gg, 128, 0, stream>>>(xb, Wqb, Qb);
  qkv_gemm_kernel<<<gg, 128, 0, stream>>>(xb, Wkb, Kb);
  qkv_gemm_kernel<<<gg, 128, 0, stream>>>(xb, Wvb, Vb);

  flash_kernel<<<dim3(S_ / 128, B_ * H_), 256, 0, stream>>>(Qb, Kb, Vb, Cb);

  out_gemm_kernel<<<gg, 128, 0, stream>>>(Cb, Wob, bo, out);
}
